// SelfAttention_51694226375380
// MI455X (gfx1250) — matrix operand order repaired and checked by reading, not compiled, not run on hardware
//
#include <hip/hip_runtime.h>
#include <hip/hip_bf16.h>

typedef __attribute__((ext_vector_type(16))) __bf16 v16bf;
typedef __attribute__((ext_vector_type(8)))  float  v8f;

#define B_SZ 4
#define S_SZ 4096
#define D_INP 1024
#define AD   512
#define OD   1024
#define EQ   2048            // qkv_dim = 2*512 + 1024
#define M_SZ (B_SZ * S_SZ)   // 16384
#define QK_W 1024            // Q(0..511) + K(512..1023) stored row-major

union FragU { uint4 q[2]; v16bf v; };

__device__ __forceinline__ unsigned short f2bf(float x) {
  unsigned int u = __float_as_uint(x);
  u += 0x7FFFu + ((u >> 16) & 1u);          // round-to-nearest-even
  return (unsigned short)(u >> 16);
}

// A-matrix 16x32 bf16 fragment: lane L -> row L&15,
// halves 0..15 = K(16h..16h+15), h = L>>4.
__device__ __forceinline__ v16bf loadFragA(const unsigned short* base, int stride, int a0) {
  int lane = threadIdx.x & 31;
  const unsigned short* p = base + (lane & 15) * stride + a0 + ((lane >> 4) << 4);
  FragU u;
  u.q[0] = *(const uint4*)(p);
  u.q[1] = *(const uint4*)(p + 8);
  return u.v;
}

// B-matrix 32x16 bf16 fragment: lane n -> column n&15,
// halves 0..15 = K(16g .. 16g+15), g = n>>4. Source row-major [N][K] -> contiguous.
__device__ __forceinline__ v16bf loadFragB(const unsigned short* base, int stride, int a0) {
  int lane = threadIdx.x & 31;
  const unsigned short* p = base + (lane & 15) * stride + a0 + ((lane >> 4) << 4);
  FragU u;
  u.q[0] = *(const uint4*)(p);
  u.q[1] = *(const uint4*)(p + 8);
  return u.v;
}

__device__ __forceinline__ v8f wmma_bf16(v16bf a, v16bf b, v8f c) {
  return __builtin_amdgcn_wmma_f32_16x16x32_bf16(false, a, false, b, (short)0, c, false, false);
}

// ---------------- K0a: X fp32 -> bf16 ----------------
__global__ __launch_bounds__(256) void k_cvt_x(const float* __restrict__ X,
                                               unsigned short* __restrict__ Xb) {
  int i = (blockIdx.x * 256 + threadIdx.x) * 4;
  float4 v = *(const float4*)(X + i);
  Xb[i + 0] = f2bf(v.x); Xb[i + 1] = f2bf(v.y);
  Xb[i + 2] = f2bf(v.z); Xb[i + 3] = f2bf(v.w);
}

// ---------------- K0b: W (D x E) fp32 -> Wt (E x D) bf16 ----------------
__global__ __launch_bounds__(256) void k_cvt_w(const float* __restrict__ W,
                                               unsigned short* __restrict__ Wt) {
  int i = blockIdx.x * 256 + threadIdx.x;      // 0 .. E*D-1
  int e = i / D_INP, d = i - e * D_INP;
  Wt[e * D_INP + d] = f2bf(W[(size_t)d * EQ + e]);
}

// ---------------- K1: QKV = Xb @ W  (bf16 WMMA, f32 acc) ----------------
// Q,K slice -> QK row-major [m][0..1023]; V slice -> Vt transposed [b][o][s].
__global__ __launch_bounds__(256) void k_qkv_gemm(const unsigned short* __restrict__ Xb,
                                                  const unsigned short* __restrict__ Wt,
                                                  unsigned short* __restrict__ QK,
                                                  unsigned short* __restrict__ Vt) {
  int wave = blockIdx.x * 8 + (threadIdx.x >> 5);
  int nt = wave & 63;            // EQ/32 = 64 tiles
  int mt = wave >> 6;            // M/32  = 512 tiles
  int m0 = mt * 32, e0 = nt * 32;
  int lane = threadIdx.x & 31, n = lane & 15, h = lane >> 4;

  v8f acc[2][2] = {};
  for (int d0 = 0; d0 < D_INP; d0 += 32) {
    v16bf a0f = loadFragA(Xb + (size_t)m0 * D_INP,        D_INP, d0);
    v16bf a1f = loadFragA(Xb + (size_t)(m0 + 16) * D_INP, D_INP, d0);
    v16bf b0f = loadFragB(Wt + (size_t)e0 * D_INP,        D_INP, d0);
    v16bf b1f = loadFragB(Wt + (size_t)(e0 + 16) * D_INP, D_INP, d0);
    acc[0][0] = wmma_bf16(a0f, b0f, acc[0][0]);
    acc[0][1] = wmma_bf16(a0f, b1f, acc[0][1]);
    acc[1][0] = wmma_bf16(a1f, b0f, acc[1][0]);
    acc[1][1] = wmma_bf16(a1f, b1f, acc[1][1]);
  }
  for (int mi = 0; mi < 2; mi++) {
    int mrow = m0 + 16 * mi;
    int b = mrow >> 12, s0 = mrow & (S_SZ - 1);
    for (int ni = 0; ni < 2; ni++) {
      int ecol = e0 + 16 * ni + n;
      if (ecol < QK_W) {                       // Q or K: row-major store
        for (int r = 0; r < 8; r++)
          QK[(size_t)(mrow + r + 8 * h) * QK_W + ecol] = f2bf(acc[mi][ni][r]);
      } else {                                 // V: transposed store -> Vt[b][o][s]
        int o = ecol - QK_W;
        union { unsigned short us[8]; uint4 q; } pk;
        for (int r = 0; r < 8; r++) pk.us[r] = f2bf(acc[mi][ni][r]);
        *(uint4*)(Vt + ((size_t)(b * OD + o)) * S_SZ + s0 + 8 * h) = pk.q;
      }
    }
  }
}

// ---------------- K2: per-key-column softmax stats (axis = q) ----------------
// m[b,k] = max_q S, Z[b,k] = sum_q exp(S - m).  Online over 16-q WMMA tiles.
__global__ __launch_bounds__(256) void k_colstats(const unsigned short* __restrict__ QK,
                                                  float* __restrict__ mOut,
                                                  float* __restrict__ zOut) {
  const float scale = 0.044194173824159216f;   // 1/sqrt(512)
  int wave = threadIdx.x >> 5, lane = threadIdx.x & 31;
  int b = blockIdx.x >> 5;                     // 128 blocks = 4 * 32
  int kbase = (blockIdx.x & 31) * 128 + wave * 16;
  const unsigned short* Qp = QK + (size_t)b * S_SZ * QK_W;
  const unsigned short* Kp = QK + ((size_t)(b * S_SZ + kbase)) * QK_W + AD;
  float mloc = -INFINITY, zloc = 0.0f;
  for (int q0 = 0; q0 < S_SZ; q0 += 16) {
    v8f acc = {};
    const unsigned short* Qrow = Qp + (size_t)q0 * QK_W;
    for (int a0 = 0; a0 < AD; a0 += 32) {
      v16bf af = loadFragA(Qrow, QK_W, a0);
      v16bf bf_ = loadFragB(Kp, QK_W, a0);
      acc = wmma_bf16(af, bf_, acc);
    }
    for (int r = 0; r < 8; r++) {
      float s = acc[r] * scale;
      float mnew = fmaxf(mloc, s);
      zloc = zloc * __expf(mloc - mnew) + __expf(s - mnew);
      mloc = mnew;
    }
  }
  float mo = __shfl_xor(mloc, 16, 32);         // lane L <-> L^16 hold same column
  float zo = __shfl_xor(zloc, 16, 32);
  float mf = fmaxf(mloc, mo);
  float zf = zloc * __expf(mloc - mf) + zo * __expf(mo - mf);
  if (lane < 16) {
    mOut[b * S_SZ + kbase + lane] = mf;
    zOut[b * S_SZ + kbase + lane] = zf;
  }
}

// ---------------- K3: out[q,:] = sum_k exp(S-m[k])/Z[k] * V[k,:] ----------------
// Block: 16 queries x full OD. 8 waves, each owns 128 output cols.
// Per 128-key chunk: wave w builds P for its 16 keys (S via WMMA from LDS-staged Q),
// writes bf16 P to LDS, then all waves consume P against Vt.
__global__ __launch_bounds__(256) void k_attn_out(const unsigned short* __restrict__ QK,
                                                  const unsigned short* __restrict__ Vt,
                                                  const float* __restrict__ mIn,
                                                  const float* __restrict__ zIn,
                                                  float* __restrict__ out) {
  __shared__ unsigned short Qs[16 * AD];       // 16KB
  __shared__ unsigned short Pbuf[16 * 128];    // 4KB
  const float scale = 0.044194173824159216f;
  int tid = threadIdx.x, wave = tid >> 5, lane = tid & 31;
  int n = lane & 15, h = lane >> 4;
  int b = blockIdx.x >> 8;                     // 1024 blocks = 4 * 256
  int q0 = (blockIdx.x & 255) * 16;

  const unsigned short* Qg = QK + ((size_t)(b * S_SZ + q0)) * QK_W;
  for (int i = tid; i < (16 * AD) / 8; i += 256) {
    int row = i >> 6, col8 = (i & 63) * 8;
    *(uint4*)&Qs[row * AD + col8] = *(const uint4*)&Qg[(size_t)row * QK_W + col8];
  }
  __syncthreads();

  int o0 = wave * 128;
  v8f acc[8] = {};
  for (int kc = 0; kc < S_SZ; kc += 128) {
    int ks = kc + wave * 16;
    v8f sacc = {};
    const unsigned short* Kp = QK + ((size_t)(b * S_SZ + ks)) * QK_W + AD;
    for (int a0 = 0; a0 < AD; a0 += 32) {
      v16bf af = loadFragA(Qs, AD, a0);        // ds_load_b128 A-fragments
      v16bf bf_ = loadFragB(Kp, QK_W, a0);
      sacc = wmma_bf16(af, bf_, sacc);
    }
    float mk = mIn[b * S_SZ + ks + n];
    float rz = 1.0f / zIn[b * S_SZ + ks + n];
    for (int r = 0; r < 8; r++) {
      float p = __expf(sacc[r] * scale - mk) * rz;
      Pbuf[(r + 8 * h) * 128 + wave * 16 + n] = f2bf(p);
    }
    __syncthreads();
    for (int c = 0; c < 4; c++) {
      v16bf pf = loadFragA(Pbuf, 128, 32 * c); // P as A-fragment from LDS
      for (int j = 0; j < 8; j++) {
        v16bf vf = loadFragB(Vt + ((size_t)(b * OD + o0 + 16 * j)) * S_SZ, S_SZ, kc + 32 * c);
        acc[j] = wmma_bf16(pf, vf, acc[j]);
      }
    }
    __syncthreads();
  }
  for (int j = 0; j < 8; j++) {
    int o = o0 + 16 * j + n;
    for (int r = 0; r < 8; r++) {
      int q = q0 + r + 8 * h;
      out[((size_t)(b * S_SZ + q)) * OD + o] = acc[j][r];
    }
  }
}

extern "C" void kernel_launch(void* const* d_in, const int* in_sizes, int n_in,
                              void* d_out, int out_size, void* d_ws, size_t ws_size,
                              hipStream_t stream) {
  const float* X = (const float*)d_in[0];      // (4,4096,1024) fp32
  const float* W = (const float*)d_in[1];      // (1024,2048)  fp32
  float* out = (float*)d_out;                  // (4,4096,1024) fp32

  char* ws = (char*)d_ws;
  unsigned short* Xb = (unsigned short*)(ws);                         // 33.5 MB
  unsigned short* Wt = (unsigned short*)(ws + 33554432);              //  4.0 MB
  unsigned short* QK = (unsigned short*)(ws + 37748736);              // 33.5 MB
  unsigned short* Vt = (unsigned short*)(ws + 71303168);              // 33.5 MB
  float*          mB = (float*)(ws + 104857600);                      // 64 KB
  float*          zB = (float*)(ws + 104923136);                      // 64 KB

  k_cvt_x   <<<dim3((M_SZ * D_INP) / 1024), dim3(256), 0, stream>>>(X, Xb);
  k_cvt_w   <<<dim3((EQ * D_INP) / 256),    dim3(256), 0, stream>>>(W, Wt);
  k_qkv_gemm<<<dim3((M_SZ / 32) * (EQ / 32) / 8), dim3(256), 0, stream>>>(Xb, Wt, QK, Vt);
  k_colstats<<<dim3(B_SZ * (S_SZ / 128)),   dim3(256), 0, stream>>>(QK, mB, zB);
  k_attn_out<<<dim3(B_SZ * (S_SZ / 16)),    dim3(256), 0, stream>>>(QK, Vt, mB, zB, out);
}
